// HeteroGNN_GraphConv_71408126263407
// MI455X (gfx1250) — matrix operand order repaired and checked by reading, not compiled, not run on hardware
//
#include <hip/hip_runtime.h>
#include <hip/hip_bf16.h>

// ---------------------------------------------------------------------------
// HeteroGNN on MI455X (gfx1250): f16 WMMA GEMMs (LDS-staged A, register-cached
// B fragments, software-pipelined fragment loads) + fp32 atomic aggregation.
// ---------------------------------------------------------------------------

typedef __attribute__((ext_vector_type(16))) _Float16 v16h;
typedef __attribute__((ext_vector_type(8)))  _Float16 v8h;
typedef __attribute__((ext_vector_type(8)))  float    v8f;

#define GNN_N     50000       // N_A == N_B
#define GNN_E     600000
#define GNN_H     128
#define GNN_B     64
#define GNN_C     10
#define MT_PER_BLK 5          // 5 M-tiles of 16 rows per block
#define ROWS_PER_BLK (MT_PER_BLK * 16)          // 80
#define GEMM_BLOCKS (GNN_N / ROWS_PER_BLK)      // 625, exact

// ---------------------------------------------------------------------------
// Weight packing: fp32 row-major W[k][n] (128x128) -> f16 WMMA B-fragment
// layout.  For each (nt, kt) 16x32 sub-block, lane l (0..31) owns column
// n = nt*16 + (l&15); lanes 0-15 hold K = kt*32 + {0..7,16..23}, lanes 16-31 hold
// K = kt*32 + {8..15,24..31} (CDNA5 ISA, 16-bit B layout).  Each lane's 16
// halves are stored contiguously -> two b128 loads per fragment.
// ---------------------------------------------------------------------------
__global__ void pack_w_kernel(const float* __restrict__ W, _Float16* __restrict__ Wp) {
    int idx = blockIdx.x * blockDim.x + threadIdx.x;  // 0..16383
    if (idx >= GNN_H * GNN_H) return;
    int k = idx >> 7, n = idx & 127;
    int nt = n >> 4, col = n & 15;
    int kt = k >> 5, kk = k & 31;
    int lane = col + ((kk & 8) ? 16 : 0);
    int h    = (kk & 7) + ((kk & 16) ? 8 : 0);
    Wp[(((nt * 4 + kt) * 32) + lane) * 16 + h] = (_Float16)W[k * GNN_H + n];
}

__global__ void f32_to_f16_kernel(const float* __restrict__ x, _Float16* __restrict__ y, int n) {
    int i = blockIdx.x * blockDim.x + threadIdx.x;
    if (i < n) y[i] = (_Float16)x[i];
}

__global__ void zero_f32_kernel(float* __restrict__ p, int n) {
    int i = blockIdx.x * blockDim.x + threadIdx.x;
    if (i < n) p[i] = 0.0f;
}

// ---------------------------------------------------------------------------
// Fragment loaders
// ---------------------------------------------------------------------------
// A (16x32 f16, MxK) from LDS: lanes 0-15 -> M=lane, K blocks {0..7,16..23};
// lanes 16-31 -> M=lane-16, K blocks {8..15,24..31}.  Two ds_load_b128.
__device__ __forceinline__ v16h load_a_frag_lds(const _Float16* sm,
                                                int mt, int kt, int lane) {
    int row = mt * 16 + (lane & 15);
    int kb  = kt * 32 + ((lane & 16) ? 8 : 0);
    const _Float16* p = sm + row * GNN_H + kb;
    v8h lo = *(const v8h*)(p);
    v8h hi = *(const v8h*)(p + 16);
    v16h a;
#pragma unroll
    for (int i = 0; i < 8; ++i) { a[i] = lo[i]; a[i + 8] = hi[i]; }
    return a;
}

// B fragment: pre-packed, lane-contiguous (see pack_w_kernel). Two b128 loads.
__device__ __forceinline__ v16h load_b_frag(const _Float16* __restrict__ wp,
                                            int nt, int kt, int lane) {
    const _Float16* p = wp + (((nt * 4 + kt) * 32) + lane) * 16;
    v8h lo = *(const v8h*)(p);
    v8h hi = *(const v8h*)(p + 8);
    v16h b;
#pragma unroll
    for (int i = 0; i < 8; ++i) { b[i] = lo[i]; b[i + 8] = hi[i]; }
    return b;
}

// ---------------------------------------------------------------------------
// out[M x 128] = act( A0 @ W0  (+ A1 @ W1)  + bias )   [f16 out, fp32 accum]
// One block = 80 output rows (5 M-tiles); 8 waves = 8 column tiles of 16.
// A tile(s) staged once in LDS (20/40 KB of the 320 KB WGP LDS); each wave
// keeps its 4 (or 8) B fragments in registers and issues 20 (or 40)
// v_wmma_f32_16x16x32_f16, with A-fragment loads software-pipelined one
// step ahead so ds_load latency overlaps the matrix op.
// ---------------------------------------------------------------------------
__global__ __launch_bounds__(256)
void gemm128_kernel(const _Float16* __restrict__ A0, const _Float16* __restrict__ W0,
                    const _Float16* __restrict__ A1, const _Float16* __restrict__ W1,
                    const float* __restrict__ bias,
                    _Float16* __restrict__ out, int M, int do_relu) {
    __shared__ _Float16 sA[2][ROWS_PER_BLK * GNN_H];   // 2 x 20 KB

    const int tid   = threadIdx.x;
    const int lane  = tid & 31;
    const int nt    = tid >> 5;                        // 0..7 column tile
    const long mbase = (long)blockIdx.x * ROWS_PER_BLK;

    // ---- cooperative stage of A tile(s) into LDS (5 x b128 per thread each)
    {
        const v8h* g0 = (const v8h*)(A0 + mbase * GNN_H);
        v8h*       s0 = (v8h*)sA[0];
#pragma unroll
        for (int j = 0; j < 5; ++j) s0[tid + j * 256] = g0[tid + j * 256];
        if (A1 != nullptr) {
            const v8h* g1 = (const v8h*)(A1 + mbase * GNN_H);
            v8h*       s1 = (v8h*)sA[1];
#pragma unroll
            for (int j = 0; j < 5; ++j) s1[tid + j * 256] = g1[tid + j * 256];
        }
    }

    // ---- B fragments -> registers (reused across all 5 M-tiles)
    v16h b0[4];
#pragma unroll
    for (int kt = 0; kt < 4; ++kt) b0[kt] = load_b_frag(W0, nt, kt, lane);
    v16h b1[4];
    if (A1 != nullptr) {
#pragma unroll
        for (int kt = 0; kt < 4; ++kt) b1[kt] = load_b_frag(W1, nt, kt, lane);
    }

    __syncthreads();

    // ---- WMMA main loops: 1-deep software pipeline on A fragments
    v8f c[MT_PER_BLK];
#pragma unroll
    for (int mt = 0; mt < MT_PER_BLK; ++mt) c[mt] = (v8f){};

    v16h a_cur = load_a_frag_lds(sA[0], 0, 0, lane);

#pragma unroll
    for (int st = 0; st < 4 * MT_PER_BLK; ++st) {
        const int mt = st >> 2, kt = st & 3;
        v16h a_nxt = a_cur;
        if (st < 4 * MT_PER_BLK - 1) {
            a_nxt = load_a_frag_lds(sA[0], (st + 1) >> 2, (st + 1) & 3, lane);
        } else if (A1 != nullptr) {
            a_nxt = load_a_frag_lds(sA[1], 0, 0, lane);
        }
        c[mt] = __builtin_amdgcn_wmma_f32_16x16x32_f16(false, a_cur, false, b0[kt],
                                                       (short)0, c[mt], false, false);
        a_cur = a_nxt;
    }
    if (A1 != nullptr) {
#pragma unroll
        for (int st = 0; st < 4 * MT_PER_BLK; ++st) {
            const int mt = st >> 2, kt = st & 3;
            v16h a_nxt = a_cur;
            if (st < 4 * MT_PER_BLK - 1)
                a_nxt = load_a_frag_lds(sA[1], (st + 1) >> 2, (st + 1) & 3, lane);
            c[mt] = __builtin_amdgcn_wmma_f32_16x16x32_f16(false, a_cur, false, b1[kt],
                                                           (short)0, c[mt], false, false);
            a_cur = a_nxt;
        }
    }

    // ---- epilogue: bias, optional ReLU, f16 store
    const int col = nt * 16 + (lane & 15);
    const float bv = bias ? bias[col] : 0.0f;
#pragma unroll
    for (int mt = 0; mt < MT_PER_BLK; ++mt) {
        const long rb = mbase + mt * 16 + ((lane & 16) ? 8 : 0);
#pragma unroll
        for (int r = 0; r < 8; ++r) {
            float v = c[mt][r] + bv;
            if (do_relu) v = v > 0.0f ? v : 0.0f;
            out[(rb + r) * GNN_H + col] = (_Float16)v;
        }
    }
    (void)M;
}

// ---------------------------------------------------------------------------
// Edge scatter-add: one wave per edge, lane covers 4 features.
// ei layout: [0..E) = src indices, [E..2E) = dst indices.
// ---------------------------------------------------------------------------
__global__ void edge_scatter_kernel(const int* __restrict__ ei,
                                    const _Float16* __restrict__ hsrc,
                                    float* __restrict__ agg, float* __restrict__ cnt,
                                    int ne) {
    int wid  = (blockIdx.x * blockDim.x + threadIdx.x) >> 5;
    int lane = threadIdx.x & 31;
    if (wid >= ne) return;
    int src = ei[wid];
    int dst = ei[ne + wid];
    const _Float16* p = hsrc + (long)src * GNN_H + lane * 4;
    float*          q = agg  + (long)dst * GNN_H + lane * 4;
#pragma unroll
    for (int j = 0; j < 4; ++j) atomicAdd(q + j, (float)p[j]);
    if (lane == 0) atomicAdd(cnt + dst, 1.0f);
}

__global__ void mean_to_f16_kernel(const float* __restrict__ agg, const float* __restrict__ cnt,
                                   _Float16* __restrict__ out, int n) {
    int i = blockIdx.x * blockDim.x + threadIdx.x;
    if (i >= n) return;
    float cc = cnt[i >> 7];
    out[i] = (_Float16)(agg[i] / fmaxf(cc, 1.0f));
}

// ---------------------------------------------------------------------------
// Segment-mean pooling: one wave per node.
// ---------------------------------------------------------------------------
__global__ void pool_scatter_kernel(const int* __restrict__ batch, const _Float16* __restrict__ h,
                                    float* __restrict__ pooled, float* __restrict__ pcnt,
                                    int n, int colofs) {
    int wid  = (blockIdx.x * blockDim.x + threadIdx.x) >> 5;
    int lane = threadIdx.x & 31;
    if (wid >= n) return;
    int b = batch[wid];
    const _Float16* p = h + (long)wid * GNN_H + lane * 4;
    float*          q = pooled + b * (2 * GNN_H) + colofs + lane * 4;
#pragma unroll
    for (int j = 0; j < 4; ++j) atomicAdd(q + j, (float)p[j]);
    if (lane == 0) atomicAdd(pcnt + b, 1.0f);
}

__global__ void pool_final_kernel(float* __restrict__ pooled, const float* __restrict__ pcnt) {
    int i = blockIdx.x * blockDim.x + threadIdx.x;
    if (i >= GNN_B * 2 * GNN_H) return;
    int b = i >> 8, c = i & 255;
    float cc = pcnt[(c < GNN_H ? 0 : GNN_B) + b];
    pooled[i] /= fmaxf(cc, 1.0f);
}

__global__ void classifier_kernel(const float* __restrict__ pooled, const float* __restrict__ Wc,
                                  const float* __restrict__ bc, float* __restrict__ out) {
    int b = blockIdx.x;
    int c = threadIdx.x;
    if (c >= GNN_C) return;
    float s = bc[c];
    for (int k = 0; k < 2 * GNN_H; ++k) s += pooled[b * 2 * GNN_H + k] * Wc[k * GNN_C + c];
    out[b * GNN_C + c] = s;
}

// ---------------------------------------------------------------------------
// Host orchestration
// ---------------------------------------------------------------------------
extern "C" void kernel_launch(void* const* d_in, const int* in_sizes, int n_in,
                              void* d_out, int out_size, void* d_ws, size_t ws_size,
                              hipStream_t stream) {
    (void)in_sizes; (void)n_in; (void)out_size; (void)ws_size;

    // ---- input mapping (setup_inputs dict order, params in insertion order)
    const float* x_a      = (const float*)d_in[0];
    const float* x_b      = (const float*)d_in[1];
    const int*   edge_ab  = (const int*)d_in[2];
    const int*   edge_ba  = (const int*)d_in[3];
    const int*   batch_a  = (const int*)d_in[4];
    const int*   batch_b  = (const int*)d_in[5];
    // d_in[6] = batch_size scalar (== 64, compile-time constant here)
    const int P = 7;
    // emb_a: W1,b1,W2,b2 ; emb_b: W1,b1,W2,b2
    // block i (base 15+i*14): Wr_ab,br_ab,Wo_ab, Wr_ba,br_ba,Wo_ba,
    //                         paW1,pab1,paW2,pab2, pbW1,pbb1,pbW2,pbb2
    // cls: Wc (43), bc (44)
    auto F = [&](int i) -> const float* { return (const float*)d_in[i]; };

    // ---- workspace bump allocator
    char* ws = (char*)d_ws;
    size_t off = 0;
    auto alloc = [&](size_t bytes) -> void* {
        void* p = ws + off;
        off += (bytes + 255) & ~(size_t)255;
        return p;
    };
    const size_t FBUF = (size_t)GNN_N * GNN_H * sizeof(_Float16);   // 12.8 MB
    _Float16* PK  = (_Float16*)alloc(20 * (size_t)GNN_H * GNN_H * sizeof(_Float16));
    _Float16* HA  = (_Float16*)alloc(FBUF);
    _Float16* HB  = (_Float16*)alloc(FBUF);
    _Float16* T1  = (_Float16*)alloc(FBUF);
    _Float16* T2  = (_Float16*)alloc(FBUF);
    _Float16* HID = (_Float16*)alloc(FBUF);   // also doubles as x16 staging
    _Float16* G   = (_Float16*)alloc(FBUF);
    float* AGG    = (float*)alloc((size_t)GNN_N * GNN_H * sizeof(float));
    float* CNT    = (float*)alloc((size_t)GNN_N * sizeof(float));
    float* POOLED = (float*)alloc((size_t)GNN_B * 2 * GNN_H * sizeof(float));
    float* PCNT   = (float*)alloc((size_t)2 * GNN_B * sizeof(float));

    auto pk = [&](int slot) -> _Float16* { return PK + (size_t)slot * GNN_H * GNN_H; };

    // ---- pack the 20 HxH weight matrices into WMMA B-fragment layout
    // slots: 0..3 emb weights; block i base 4+i*8:
    //   +0 Wr_ab, +1 Wo_ab, +2 Wr_ba, +3 Wo_ba, +4 paW1, +5 paW2, +6 pbW1, +7 pbW2
    const int widx[20] = {
        P + 0, P + 2, P + 4, P + 6,                               // emb_a W1,W2 / emb_b W1,W2
        15 + 0, 15 + 2, 15 + 3, 15 + 5, 15 + 6, 15 + 8, 15 + 10, 15 + 12,   // block 0
        29 + 0, 29 + 2, 29 + 3, 29 + 5, 29 + 6, 29 + 8, 29 + 10, 29 + 12    // block 1
    };
    for (int s = 0; s < 20; ++s)
        pack_w_kernel<<<64, 256, 0, stream>>>(F(widx[s]), pk(s));

    const int NH = GNN_N * GNN_H;
    const int EB = (GNN_E * 32 + 255) / 256;   // blocks for 1 wave/edge
    const int NB = (GNN_N * 32 + 255) / 256;   // blocks for 1 wave/node

    // ---- embedding MLPs (relu(relu(x@W1+b1)@W2+b2))
    f32_to_f16_kernel<<<(NH + 255) / 256, 256, 0, stream>>>(x_a, HID, NH);
    gemm128_kernel<<<GEMM_BLOCKS, 256, 0, stream>>>(HID, pk(0), nullptr, nullptr, F(P + 1), T1, GNN_N, 1);
    gemm128_kernel<<<GEMM_BLOCKS, 256, 0, stream>>>(T1,  pk(1), nullptr, nullptr, F(P + 3), HA, GNN_N, 1);
    f32_to_f16_kernel<<<(NH + 255) / 256, 256, 0, stream>>>(x_b, HID, NH);
    gemm128_kernel<<<GEMM_BLOCKS, 256, 0, stream>>>(HID, pk(2), nullptr, nullptr, F(P + 5), T1, GNN_N, 1);
    gemm128_kernel<<<GEMM_BLOCKS, 256, 0, stream>>>(T1,  pk(3), nullptr, nullptr, F(P + 7), HB, GNN_N, 1);

    // ---- conv blocks
    for (int blk = 0; blk < 2; ++blk) {
        const int base = 15 + blk * 14;
        const int s    = 4 + blk * 8;

        // new_b = mean_{a->b}(h_a) @ Wr_ab + br_ab + h_b @ Wo_ab
        zero_f32_kernel<<<(NH + 255) / 256, 256, 0, stream>>>(AGG, NH);
        zero_f32_kernel<<<(GNN_N + 255) / 256, 256, 0, stream>>>(CNT, GNN_N);
        edge_scatter_kernel<<<EB, 256, 0, stream>>>(edge_ab, HA, AGG, CNT, GNN_E);
        mean_to_f16_kernel<<<(NH + 255) / 256, 256, 0, stream>>>(AGG, CNT, G, NH);
        gemm128_kernel<<<GEMM_BLOCKS, 256, 0, stream>>>(G, pk(s + 0), HB, pk(s + 1),
                                                        F(base + 1), T2, GNN_N, 0);

        // new_a = mean_{b->a}(h_b) @ Wr_ba + br_ba + h_a @ Wo_ba
        zero_f32_kernel<<<(NH + 255) / 256, 256, 0, stream>>>(AGG, NH);
        zero_f32_kernel<<<(GNN_N + 255) / 256, 256, 0, stream>>>(CNT, GNN_N);
        edge_scatter_kernel<<<EB, 256, 0, stream>>>(edge_ba, HB, AGG, CNT, GNN_E);
        mean_to_f16_kernel<<<(NH + 255) / 256, 256, 0, stream>>>(AGG, CNT, G, NH);
        gemm128_kernel<<<GEMM_BLOCKS, 256, 0, stream>>>(G, pk(s + 2), HA, pk(s + 3),
                                                        F(base + 4), T1, GNN_N, 0);

        // post MLPs
        gemm128_kernel<<<GEMM_BLOCKS, 256, 0, stream>>>(T1,  pk(s + 4), nullptr, nullptr, F(base + 7),  HID, GNN_N, 1);
        gemm128_kernel<<<GEMM_BLOCKS, 256, 0, stream>>>(HID, pk(s + 5), nullptr, nullptr, F(base + 9),  HA,  GNN_N, 1);
        gemm128_kernel<<<GEMM_BLOCKS, 256, 0, stream>>>(T2,  pk(s + 6), nullptr, nullptr, F(base + 11), HID, GNN_N, 1);
        gemm128_kernel<<<GEMM_BLOCKS, 256, 0, stream>>>(HID, pk(s + 7), nullptr, nullptr, F(base + 13), HB,  GNN_N, 1);
    }

    // ---- segment-mean pooling + classifier
    zero_f32_kernel<<<(GNN_B * 2 * GNN_H + 255) / 256, 256, 0, stream>>>(POOLED, GNN_B * 2 * GNN_H);
    zero_f32_kernel<<<1, 256, 0, stream>>>(PCNT, 2 * GNN_B);
    pool_scatter_kernel<<<NB, 256, 0, stream>>>(batch_a, HA, POOLED, PCNT,         GNN_N, 0);
    pool_scatter_kernel<<<NB, 256, 0, stream>>>(batch_b, HB, POOLED, PCNT + GNN_B, GNN_N, GNN_H);
    pool_final_kernel<<<(GNN_B * 2 * GNN_H + 255) / 256, 256, 0, stream>>>(POOLED, PCNT);
    classifier_kernel<<<GNN_B, 32, 0, stream>>>(POOLED, F(43), F(44), (float*)d_out);
}
